// NeuralODE_85830626443530
// MI455X (gfx1250) — matrix operand order repaired and checked by reading, not compiled, not run on hardware
//
#include <hip/hip_runtime.h>

// ---------------------------------------------------------------------------
// NeuralODE RK4 (8 steps) on MI455X (gfx1250, wave32, WMMA bf16)
//   f(z) = tanh(z@W1 + b1) @ W2 + b2
//
// GEMM1: hid = tanh(zeff @ W1 + b1)   [1024x256]x[256x1024], bf16 in/out,
//        2x4 tile blocking: 12 b128 loads -> 8 v_wmma per K-step,
//        hardware v_tanh_f32 epilogue.
// GEMM2: k = hid @ W2 + b2            [1024x1024]x[1024x256], 2x2 blocking,
//        fused epilogue: zacc += (w/6)*h*k and zeff_next = bf16(z + a*h*k)
//        (last stage: zeff_next = bf16(z_new)) -> GEMM1 never converts.
// Weights converted once per launch to bf16, N-major (B fragments = two
// contiguous global_load_b128 per lane).
// ---------------------------------------------------------------------------

typedef __attribute__((ext_vector_type(16))) __bf16 v16bf;
typedef __attribute__((ext_vector_type(8)))  __bf16 v8bf;
typedef __attribute__((ext_vector_type(4)))  __bf16 v4bf;
typedef __attribute__((ext_vector_type(8)))  float  v8f;

#define BATCH 1024
#define DIMZ  256
#define DIMH  1024
#define NSTEP 8

#if __has_builtin(__builtin_amdgcn_tanhf)
__device__ __forceinline__ float fast_tanh(float x) { return __builtin_amdgcn_tanhf(x); }
#else
__device__ __forceinline__ float fast_tanh(float x) { return tanhf(x); }
#endif

union FragBF {
    v16bf v;
    v8bf  h[2];
};

__device__ __forceinline__ v8f wmma_bf16(const FragBF& a, const FragBF& b, v8f c) {
    return __builtin_amdgcn_wmma_f32_16x16x32_bf16(
        false, a.v, false, b.v, (short)0, c, false, false);
}

// --- weight convert + transpose: src[k*N+n] (f32) -> dst[n*K+k] (bf16) -----
template <int K, int N>
__global__ void wt_convert(const float* __restrict__ src,
                           __bf16* __restrict__ dst) {
    const int idx = blockIdx.x * blockDim.x + threadIdx.x;   // over K*N
    const int k = idx / N;                                   // N = 2^p -> shift
    const int n = idx & (N - 1);
    dst[n * K + k] = (__bf16)src[idx];
}

// --- z0 (f32) -> zeff (bf16), 4 elements/thread ----------------------------
__global__ void z2bf_kernel(const float* __restrict__ z,
                            __bf16* __restrict__ out) {
    const int idx = (blockIdx.x * blockDim.x + threadIdx.x) * 4;
    float4 v = *(const float4*)(z + idx);
    v4bf o;
    o[0] = (__bf16)v.x; o[1] = (__bf16)v.y;
    o[2] = (__bf16)v.z; o[3] = (__bf16)v.w;
    *(v4bf*)(out + idx) = o;
}

// --- GEMM1: hid = tanh(zeff @ W1 + b1), pure bf16 WMMA, 2x4 blocking -------
// 32x64 output per wave; K=256 -> 8 K-steps x 8 wmma
__global__ __launch_bounds__(256)
void gemm1_kernel(const __bf16* __restrict__ zeff,
                  const __bf16* __restrict__ w1t,
                  const float* __restrict__ bias1,
                  __bf16* __restrict__ hid) {
    const int lane = threadIdx.x & 31;
    const int wid  = blockIdx.x * 8 + (threadIdx.x >> 5);
    const int mblk = wid >> 4;                 // 32 M blocks of 32 rows
    const int nblk = wid & 15;                 // 16 N blocks of 64 cols
    const int ln   = lane & 15;
    const int half = lane >> 4;
    const int aoff = half * 8;
    const int boff = half * 8;

    const __bf16* ap[2];
    ap[0] = zeff + (mblk * 32 + ln) * DIMZ;
    ap[1] = ap[0] + 16 * DIMZ;
    const __bf16* bp[4];
#pragma unroll
    for (int j = 0; j < 4; ++j)
        bp[j] = w1t + (nblk * 64 + j * 16 + ln) * DIMZ;      // N-major

    v8f acc[2][4];
#pragma unroll
    for (int i = 0; i < 2; ++i)
#pragma unroll
        for (int j = 0; j < 4; ++j) acc[i][j] = (v8f){};

#pragma unroll
    for (int ks = 0; ks < DIMZ / 32; ++ks) {
        const int kb = ks * 32;
        FragBF fa[2], fb[4];
#pragma unroll
        for (int i = 0; i < 2; ++i) {
            fa[i].h[0] = *(const v8bf*)(ap[i] + kb + aoff);
            fa[i].h[1] = *(const v8bf*)(ap[i] + kb + 16 + aoff);
        }
#pragma unroll
        for (int j = 0; j < 4; ++j) {
            fb[j].h[0] = *(const v8bf*)(bp[j] + kb + boff);
            fb[j].h[1] = *(const v8bf*)(bp[j] + kb + 16 + boff);
        }
#pragma unroll
        for (int i = 0; i < 2; ++i)
#pragma unroll
            for (int j = 0; j < 4; ++j)
                acc[i][j] = wmma_bf16(fa[i], fb[j], acc[i][j]);
    }

    const int m0 = mblk * 32 + half * 8;
#pragma unroll
    for (int j = 0; j < 4; ++j) {
        const int ncol = nblk * 64 + j * 16 + ln;
        const float bv = bias1[ncol];
#pragma unroll
        for (int i = 0; i < 2; ++i) {
#pragma unroll
            for (int r = 0; r < 8; ++r) {
                const int m = m0 + i * 16 + r;
                hid[m * DIMH + ncol] = (__bf16)fast_tanh(acc[i][j][r] + bv);
            }
        }
    }
}

// --- GEMM2: k = hid @ W2 + b2; fused RK4 accumulate + next-stage A-build ---
// 32x32 output per wave; K=1024 -> 32 K-steps x 4 wmma
__global__ __launch_bounds__(256)
void gemm2_kernel(const __bf16* __restrict__ hid,
                  const __bf16* __restrict__ w2t,
                  const float* __restrict__ bias2,
                  const float* __restrict__ tptr,
                  float cW, float cAnext, int initFlag, int lastFlag,
                  const float* __restrict__ zin,
                  float* __restrict__ zacc,
                  __bf16* __restrict__ zeffout) {
    const int lane = threadIdx.x & 31;
    const int wid  = blockIdx.x * 8 + (threadIdx.x >> 5);
    const int mblk = wid >> 3;                 // 32 M blocks of 32 rows
    const int nblk = wid & 7;                  // 8  N blocks of 32 cols
    const int ln   = lane & 15;
    const int half = lane >> 4;
    const int aoff = half * 8;
    const int boff = half * 8;

    const __bf16* a0p = hid + (mblk * 32 + ln) * DIMH;       // row-major
    const __bf16* a1p = a0p + 16 * DIMH;
    const __bf16* b0p = w2t + (nblk * 32 + ln) * DIMH;       // N-major
    const __bf16* b1p = b0p + 16 * DIMH;

    v8f acc00 = {}, acc01 = {}, acc10 = {}, acc11 = {};
#pragma unroll 4
    for (int ks = 0; ks < DIMH / 32; ++ks) {
        const int kb = ks * 32;
        FragBF fa0, fa1, fb0, fb1;
        fa0.h[0] = *(const v8bf*)(a0p + kb + aoff);
        fa0.h[1] = *(const v8bf*)(a0p + kb + 16 + aoff);
        fa1.h[0] = *(const v8bf*)(a1p + kb + aoff);
        fa1.h[1] = *(const v8bf*)(a1p + kb + 16 + aoff);
        fb0.h[0] = *(const v8bf*)(b0p + kb + boff);
        fb0.h[1] = *(const v8bf*)(b0p + kb + 16 + boff);
        fb1.h[0] = *(const v8bf*)(b1p + kb + boff);
        fb1.h[1] = *(const v8bf*)(b1p + kb + 16 + boff);
        acc00 = wmma_bf16(fa0, fb0, acc00);
        acc01 = wmma_bf16(fa0, fb1, acc01);
        acc10 = wmma_bf16(fa1, fb0, acc10);
        acc11 = wmma_bf16(fa1, fb1, acc11);
    }

    const float h  = (tptr[1] - tptr[0]) * (1.0f / NSTEP);
    const float wc = cW * h;       // RK4 combine weight (w/6 folded on host)
    const float an = cAnext * h;   // next-stage axpy coefficient

    const int n0 = nblk * 32 + ln, n1 = n0 + 16;
    const float bv[2] = {bias2[n0], bias2[n1]};
    const int m0 = mblk * 32 + half * 8;

#pragma unroll
    for (int r = 0; r < 8; ++r) {
#pragma unroll
        for (int i = 0; i < 2; ++i) {      // M sub-tile
#pragma unroll
            for (int j = 0; j < 2; ++j) {  // N sub-tile
                float kv;
                if (i == 0 && j == 0) kv = acc00[r];
                else if (i == 0)      kv = acc01[r];
                else if (j == 0)      kv = acc10[r];
                else                  kv = acc11[r];
                kv += bv[j];
                const int m   = m0 + i * 16 + r;
                const int n   = (j == 0) ? n0 : n1;
                const int idx = m * DIMZ + n;
                const float zb   = zin[idx];               // step-base z
                const float base = initFlag ? zb : zacc[idx];
                const float zn   = fmaf(wc, kv, base);     // RK4 accumulate
                zacc[idx] = zn;
                const float ze = lastFlag ? zn : fmaf(an, kv, zb);
                zeffout[idx] = (__bf16)ze;                 // next A operand
            }
        }
    }
}

// ---------------------------------------------------------------------------
extern "C" void kernel_launch(void* const* d_in, const int* in_sizes, int n_in,
                              void* d_out, int out_size, void* d_ws, size_t ws_size,
                              hipStream_t stream) {
    const float* z0 = (const float*)d_in[0];
    const float* t  = (const float*)d_in[1];
    const float* W1 = (const float*)d_in[2];
    const float* b1 = (const float*)d_in[3];
    const float* W2 = (const float*)d_in[4];
    const float* b2 = (const float*)d_in[5];

    char* ws = (char*)d_ws;
    __bf16* w1t  = (__bf16*)(ws);                 // 512 KB
    __bf16* w2t  = (__bf16*)(ws + (512u << 10));  // 512 KB
    __bf16* hid  = (__bf16*)(ws + (1024u << 10)); // 2 MB
    __bf16* zeff = (__bf16*)(ws + (3072u << 10)); // 512 KB
    float*  zA   = (float*)(ws + (3584u << 10));  // 1 MB
    float*  zB   = (float*)(ws + (4608u << 10));  // 1 MB

    // per-launch weight convert + transpose (L2-resident afterwards)
    wt_convert<DIMZ, DIMH><<<DIMZ * DIMH / 256, 256, 0, stream>>>(W1, w1t);
    wt_convert<DIMH, DIMZ><<<DIMH * DIMZ / 256, 256, 0, stream>>>(W2, w2t);
    // initial A operand: zeff = bf16(z0)
    z2bf_kernel<<<BATCH * DIMZ / 4 / 256, 256, 0, stream>>>(z0, zeff);

    static const float cA[4] = {0.0f, 0.5f, 0.5f, 1.0f};          // alpha / h
    static const float cW[4] = {1.0f/6.0f, 2.0f/6.0f, 2.0f/6.0f, 1.0f/6.0f};

    const int g1 = (BATCH / 32) * (DIMH / 64) / 8;  // 64 blocks, 8 waves each
    const int g2 = (BATCH / 32) * (DIMZ / 32) / 8;  // 32 blocks

    const float* zcur = z0;
    for (int step = 0; step < NSTEP; ++step) {
        float* znext = (step == NSTEP - 1) ? (float*)d_out
                                           : ((step & 1) ? zB : zA);
        for (int s = 0; s < 4; ++s) {
            gemm1_kernel<<<g1, 256, 0, stream>>>(zeff, w1t, b1, hid);
            const float cAn = (s < 3) ? cA[s + 1] : 0.0f;
            gemm2_kernel<<<g2, 256, 0, stream>>>(hid, w2t, b2, t, cW[s], cAn,
                                                 (s == 0), (s == 3),
                                                 zcur, znext, zeff);
        }
        zcur = znext;
    }
}
